// GCN_2LAYERS_SUM_58033598103990
// MI455X (gfx1250) — compile-verified, matrix operand order checked by reading, not run on hardware
//
#include <hip/hip_runtime.h>
#include <hip/hip_bf16.h>

typedef _Float16 v16h __attribute__((ext_vector_type(16)));
typedef _Float16 v4h  __attribute__((ext_vector_type(4)));
typedef float    v8f  __attribute__((ext_vector_type(8)));
typedef unsigned int u32x4 __attribute__((ext_vector_type(4)));
typedef int          i32x8 __attribute__((ext_vector_type(8)));
typedef int          i32x4 __attribute__((ext_vector_type(4)));

#define HD 128          // hidden width (=F=H in reference)
#define LDA 136         // LDS row stride in halves (128+8 pad -> bank-conflict free)

// ---------------------------------------------------------------------------
// Transpose a 128x128 f32 weight matrix into [out_col][k] f16 layout so the
// WMMA B-fragment can be loaded with the same K-striping as the A-fragment.
// ---------------------------------------------------------------------------
__global__ void __launch_bounds__(256) transpose_to_f16(
    const float* __restrict__ w, _Float16* __restrict__ wt) {
  int idx = blockIdx.x * 256 + threadIdx.x;   // idx = k*128 + c
  if (idx >= HD * HD) return;
  int k = idx >> 7;
  int c = idx & 127;
  wt[c * HD + k] = (_Float16)w[idx];
}

// ---------------------------------------------------------------------------
// C[M x 128] = A[M x 128] @ W[128 x 128] (+ bias), f16 inputs, f32 accum.
// Block: 256 threads (8 wave32) -> 128x128 output tile.
//  - W tile (f16, 32KB) staged into LDS by the Tensor Data Mover with
//    pad_enable reproducing the 136-half padded row stride.
//  - A tile staged by all 256 threads with on-the-fly f32->f16 conversion
//    (overlaps the TDM DMA).
// Wave w computes rows [16w, 16w+16), all 128 columns (8 WMMA n-tiles),
// K swept in 4 steps of 32 with v_wmma_f32_16x16x32_f16.
// ---------------------------------------------------------------------------
__global__ void __launch_bounds__(256) gemm_wmma(
    const float* __restrict__ A, const _Float16* __restrict__ Wt,
    const float* __restrict__ bias, float* __restrict__ out, int M) {
  __shared__ _Float16 lA[128 * LDA];
  __shared__ _Float16 lW[128 * LDA];

  const int tid  = threadIdx.x;
  const int wave = tid >> 5;
  const int lane = tid & 31;
  const long long rowBase = (long long)blockIdx.x * 128;

  // ---- Stage Wt (128x128 f16 tensor, row-major, stride 128) via TDM ------
  // D# group0: count=1 | lds_addr | global_addr | type=2
  // D# group1: data_size=2B, pad_enable, pad_interval=64 DW, pad_amount=4 DW
  //            tensor_dim0=128, tensor_dim1=128, tile_dim0=128, tile_dim1=128
  //            tensor_dim0_stride=128  (all in data_size units)
  if (wave == 0) {
    unsigned lds_off = (unsigned)(unsigned long long)(void*)&lW[0];
    unsigned long long ga = (unsigned long long)Wt;
    u32x4 g0 = { 1u,                                   // count=1 (valid D#)
                 lds_off,                              // bits 63:32  lds_addr
                 (unsigned)ga,                         // bits 95:64  gaddr lo
                 (unsigned)((ga >> 32) & 0x01FFFFFFu)  // bits 120:96 gaddr hi
                   | 0x80000000u };                    // bits 127:126 type=2
    i32x8 g1 = { (int)((1u << 16)        // data_size = 1 -> 2 bytes
                     | (1u << 20)        // pad_enable
                     | (5u << 22)        // pad_interval: 1<<5 * 8B = 64 DWORDs
                     | (3u << 25)),      // pad_amount: 4 DWORDs (16B)
                 (int)(128u << 16),      // tensor_dim0[15:0] @ bits 63:48
                 (int)(128u << 16),      // tensor_dim1[15:0] @ bits 95:80
                 (int)(128u << 16),      // tile_dim0        @ bits 127:112
                 128,                    // tile_dim1        @ bits 143:128
                 128,                    // tensor_dim0_stride[31:0]
                 0, 0 };                 // stride hi, tensor_dim1_stride
    i32x4 gz  = { 0, 0, 0, 0 };          // groups 2/3 unused (2D tile)
    i32x8 gz8 = { 0, 0, 0, 0, 0, 0, 0, 0 };
    __builtin_amdgcn_tensor_load_to_lds(g0, g1, gz, gz, gz8, 0);
    __builtin_amdgcn_s_wait_tensorcnt(0);   // wave0: DMA complete
  }

  // ---- Stage A tile (128 rows x 128 f32), convert to f16: 4096 float4 ----
  for (int i = tid; i < 4096; i += 256) {
    int r  = i >> 5;            // 32 float4 per row
    int c4 = (i & 31) * 4;
    long long gr = rowBase + r;
    float4 v = make_float4(0.f, 0.f, 0.f, 0.f);
    if (gr < M) v = *(const float4*)(A + gr * HD + c4);
    v4h h = { (_Float16)v.x, (_Float16)v.y, (_Float16)v.z, (_Float16)v.w };
    *(v4h*)(&lA[r * LDA + c4]) = h;
  }
  __syncthreads();              // publishes both TDM tile and A tile

  const int half_ = lane >> 4;        // which 16-lane half of the wave
  const int l16   = lane & 15;
  const int rA    = wave * 16 + l16;  // A row within the block tile

  v8f acc[8] = {};

  for (int k0 = 0; k0 < 128; k0 += 32) {
    // A fragment: ISA 16-bit A 16x32 layout.
    // element j (=2v+p): K = k0 + half*8 + j + (j>=8 ? 8 : 0)
    v16h afrag;
#pragma unroll
    for (int j = 0; j < 16; j++) {
      int k = k0 + half_ * 8 + j + ((j >> 3) << 3);
      afrag[j] = lA[rA * LDA + k];
    }
#pragma unroll
    for (int t = 0; t < 8; t++) {
      int c = t * 16 + l16;           // B column owned by this lane
      v16h bfrag;
#pragma unroll
      for (int j = 0; j < 16; j++) {
        int k = k0 + half_ * 8 + j + ((j >> 3) << 3);
        bfrag[j] = lW[c * LDA + k];
      }
      acc[t] = __builtin_amdgcn_wmma_f32_16x16x32_f16(
          false, afrag, false, bfrag, (short)0, acc[t], false, false);
    }
  }

  // D layout: lanes 0-15 -> M = j, lanes 16-31 -> M = 8 + j (N = l16).
#pragma unroll
  for (int t = 0; t < 8; t++) {
    int c = t * 16 + l16;
    float bv = bias ? bias[c] : 0.0f;
#pragma unroll
    for (int j = 0; j < 8; j++) {
      long long r = rowBase + wave * 16 + half_ * 8 + j;
      if (r < M) out[r * HD + c] = acc[t][j] + bv;
    }
  }
}

// ---------------------------------------------------------------------------
// agg[n][f] = bias[f]  (seed buffer before scatter-add)
// ---------------------------------------------------------------------------
__global__ void __launch_bounds__(256) init_bias(
    float* __restrict__ agg, const float* __restrict__ b, long long nf) {
  long long idx = (long long)blockIdx.x * 256 + threadIdx.x;
  if (idx < nf) agg[idx] = b[idx & (HD - 1)];
}

// ---------------------------------------------------------------------------
// Sum-aggregate: one wave32 per edge, 4 floats per lane (float4 gather,
// 4x global_atomic_add_f32). Edges e >= E are the synthesized self loops.
// Atomics resolve in the 192MB L2 (working set is L2-resident).
// ---------------------------------------------------------------------------
__global__ void __launch_bounds__(256) scatter_add(
    const float* __restrict__ m, const long long* __restrict__ ei,
    float* __restrict__ agg, int E, int Nn) {
  long long g = (long long)blockIdx.x * 256 + threadIdx.x;
  long long totalLanes = (long long)(E + Nn) * 32;
  if (g >= totalLanes) return;          // uniform per-wave exit
  int lane = (int)(g & 31);
  long long e = g >> 5;
  long long s, d;
  if (e < E) { s = ei[e]; d = ei[(long long)E + e]; }
  else       { s = d = e - E; }
  float4 v = *(const float4*)(m + s * HD + lane * 4);
  float* dst = agg + d * HD + lane * 4;
  atomicAdd(dst + 0, v.x);
  atomicAdd(dst + 1, v.y);
  atomicAdd(dst + 2, v.z);
  atomicAdd(dst + 3, v.w);
}

// ---------------------------------------------------------------------------
// In-place tanh
// ---------------------------------------------------------------------------
__global__ void __launch_bounds__(256) tanh_ip(float* __restrict__ p, long long nf) {
  long long idx = (long long)blockIdx.x * 256 + threadIdx.x;
  if (idx < nf) p[idx] = tanhf(p[idx]);
}

// ---------------------------------------------------------------------------
// Head: h2 = tanh(h @ cp_w + cp_b) [N,2]; out = h2 @ cls_w + cls_b [N,C].
// One wave32 per node: lane owns 4 features, wave32 shfl reduction.
// ---------------------------------------------------------------------------
__global__ void __launch_bounds__(256) head_kernel(
    const float* __restrict__ h, const float* __restrict__ cp_w,
    const float* __restrict__ cp_b, const float* __restrict__ cls_w,
    const float* __restrict__ cls_b, float* __restrict__ out,
    float* __restrict__ hout, int Nn, int C) {
  int lane = threadIdx.x & 31;
  long long node = (long long)blockIdx.x * 8 + (threadIdx.x >> 5);
  if (node >= Nn) return;
  float4 hv = *(const float4*)(h + node * HD + lane * 4);
  float d0 = 0.f, d1 = 0.f;
  float xs[4] = { hv.x, hv.y, hv.z, hv.w };
#pragma unroll
  for (int i = 0; i < 4; i++) {
    int k = lane * 4 + i;
    d0 += xs[i] * cp_w[k * 2 + 0];
    d1 += xs[i] * cp_w[k * 2 + 1];
  }
#pragma unroll
  for (int off = 16; off > 0; off >>= 1) {   // wave32 reduction
    d0 += __shfl_down(d0, off, 32);
    d1 += __shfl_down(d1, off, 32);
  }
  if (lane == 0) {
    float h0 = tanhf(d0 + cp_b[0]);
    float h1 = tanhf(d1 + cp_b[1]);
    hout[node * 2 + 0] = h0;
    hout[node * 2 + 1] = h1;
    for (int c = 0; c < C; c++)
      out[node * C + c] = h0 * cls_w[0 * C + c] + h1 * cls_w[1 * C + c] + cls_b[c];
  }
}

// ---------------------------------------------------------------------------
extern "C" void kernel_launch(void* const* d_in, const int* in_sizes, int n_in,
                              void* d_out, int out_size, void* d_ws, size_t ws_size,
                              hipStream_t stream) {
  const float*     x       = (const float*)d_in[0];
  const long long* ei      = (const long long*)d_in[1];   // int64 edge_index
  const float*     first_w = (const float*)d_in[2];
  const float*     first_b = (const float*)d_in[3];
  const float*     w1      = (const float*)d_in[4];
  const float*     b1      = (const float*)d_in[5];
  const float*     w2      = (const float*)d_in[6];
  const float*     b2      = (const float*)d_in[7];
  const float*     cp_w    = (const float*)d_in[8];
  const float*     cp_b    = (const float*)d_in[9];
  const float*     cls_w   = (const float*)d_in[10];
  const float*     cls_b   = (const float*)d_in[11];

  const int Nn = in_sizes[0] / HD;       // 50000
  const int E  = in_sizes[1] / 2;        // 600000
  const int C  = in_sizes[11];           // 4

  // Workspace layout: two N x 128 f32 ping-pong buffers + 3 f16 weight mats.
  float* bufA = (float*)d_ws;
  float* bufB = bufA + (size_t)Nn * HD;
  _Float16* wtF = (_Float16*)(bufB + (size_t)Nn * HD);
  _Float16* wt1 = wtF + HD * HD;
  _Float16* wt2 = wt1 + HD * HD;

  transpose_to_f16<<<(HD * HD + 255) / 256, 256, 0, stream>>>(first_w, wtF);
  transpose_to_f16<<<(HD * HD + 255) / 256, 256, 0, stream>>>(w1, wt1);
  transpose_to_f16<<<(HD * HD + 255) / 256, 256, 0, stream>>>(w2, wt2);

  const int gBlocks = (Nn + 127) / 128;
  const long long nf = (long long)Nn * HD;
  const int ibBlocks = (int)((nf + 255) / 256);
  const long long lanesTotal = (long long)(E + Nn) * 32;
  const int scBlocks = (int)((lanesTotal + 255) / 256);

  // h0 = x @ first_w + first_b
  gemm_wmma<<<gBlocks, 256, 0, stream>>>(x, wtF, first_b, bufA, Nn);

  // conv1: m = h0 @ w1 ; agg = b1 + scatter_sum ; h1 = tanh(agg)
  gemm_wmma<<<gBlocks, 256, 0, stream>>>(bufA, wt1, nullptr, bufB, Nn);
  init_bias<<<ibBlocks, 256, 0, stream>>>(bufA, b1, nf);
  scatter_add<<<scBlocks, 256, 0, stream>>>(bufB, ei, bufA, E, Nn);
  tanh_ip<<<ibBlocks, 256, 0, stream>>>(bufA, nf);

  // conv2
  gemm_wmma<<<gBlocks, 256, 0, stream>>>(bufA, wt2, nullptr, bufB, Nn);
  init_bias<<<ibBlocks, 256, 0, stream>>>(bufA, b2, nf);
  scatter_add<<<scBlocks, 256, 0, stream>>>(bufB, ei, bufA, E, Nn);
  tanh_ip<<<ibBlocks, 256, 0, stream>>>(bufA, nf);

  // head: writes tuple (out [N,C], h2 [N,2]) concatenated into d_out
  float* outp  = (float*)d_out;
  float* houtp = outp + (size_t)Nn * C;
  head_kernel<<<(Nn + 7) / 8, 256, 0, stream>>>(bufA, cp_w, cp_b, cls_w, cls_b,
                                                outp, houtp, Nn, C);
}
